// Part_GCN_42666205118972
// MI455X (gfx1250) — matrix operand order repaired and checked by reading, not compiled, not run on hardware
//
#include <hip/hip_runtime.h>

#define Nn 20000
#define Ee 160000
#define Hh 512
#define H2 1024
#define H3 1536
#define MSG_EPS 1e-7f

typedef __attribute__((ext_vector_type(16))) __bf16 v16bf;
typedef __attribute__((ext_vector_type(8)))  float  v8f;

union FragBF { v16bf v; uint4 q[2]; };

#define WMMA_BF16(a, b, c) \
    __builtin_amdgcn_wmma_f32_16x16x32_bf16(false, (a), false, (b), (short)0, (c), false, false)

// ---------- monotonic float<->uint encoding for atomic max ----------
__device__ __forceinline__ unsigned encf(float x) {
    unsigned u = __float_as_uint(x);
    return (u >> 31) ? ~u : (u | 0x80000000u);
}
__device__ __forceinline__ float decf(unsigned u) {
    return (u >> 31) ? __uint_as_float(u & 0x7FFFFFFFu) : __uint_as_float(~u);
}

// ---------- init zmax / num / den ----------
__global__ void k_init_seg(unsigned* __restrict__ zmax, float* __restrict__ num,
                           float* __restrict__ den, int n) {
    int i = blockIdx.x * blockDim.x + threadIdx.x;
    if (i < n) { zmax[i] = 0x007FFFFFu; /* enc(-inf) */ num[i] = 0.f; den[i] = 0.f; }
}

// ---------- pass 1: segment max of z = (relu(x[src])+eps)*t ----------
__global__ void k_edge_max(const float4* __restrict__ x, const int* __restrict__ ei,
                           const float* __restrict__ tp, unsigned* __restrict__ zmax,
                           int E_, int Hq) {
    int tid = blockIdx.x * blockDim.x + threadIdx.x;
    int e = tid / Hq, q = tid % Hq;
    if (e >= E_) return;
    int s = ei[e], d = ei[E_ + e];
    float t = tp[0];
    float4 v = x[(long)s * Hq + q];
    unsigned* zm = zmax + (long)d * (Hq * 4) + q * 4;
    atomicMax(&zm[0], encf((fmaxf(v.x, 0.f) + MSG_EPS) * t));
    atomicMax(&zm[1], encf((fmaxf(v.y, 0.f) + MSG_EPS) * t));
    atomicMax(&zm[2], encf((fmaxf(v.z, 0.f) + MSG_EPS) * t));
    atomicMax(&zm[3], encf((fmaxf(v.w, 0.f) + MSG_EPS) * t));
}

// ---------- decode zmax in place, empty segments (-inf) -> 0 ----------
__global__ void k_zmax_fix(unsigned* __restrict__ zm, int n) {
    int i = blockIdx.x * blockDim.x + threadIdx.x;
    if (i >= n) return;
    float z = decf(zm[i]);
    if (!isfinite(z)) z = 0.f;
    zm[i] = __float_as_uint(z);
}

// ---------- pass 2: num += e*m, den += e with e = exp(z - zmax[dst]) ----------
__global__ void k_edge_sum(const float4* __restrict__ x, const int* __restrict__ ei,
                           const float* __restrict__ tp, const float* __restrict__ zmax,
                           float* __restrict__ num, float* __restrict__ den,
                           int E_, int Hq) {
    int tid = blockIdx.x * blockDim.x + threadIdx.x;
    int e = tid / Hq, q = tid % Hq;
    if (e >= E_) return;
    int s = ei[e], d = ei[E_ + e];
    float t = tp[0];
    float4 v = x[(long)s * Hq + q];
    long base = (long)d * (Hq * 4) + q * 4;
    float m[4] = { fmaxf(v.x, 0.f) + MSG_EPS, fmaxf(v.y, 0.f) + MSG_EPS,
                   fmaxf(v.z, 0.f) + MSG_EPS, fmaxf(v.w, 0.f) + MSG_EPS };
#pragma unroll
    for (int j = 0; j < 4; ++j) {
        float ex = __expf(m[j] * t - zmax[base + j]);
        atomicAdd(&num[base + j], ex * m[j]);
        atomicAdd(&den[base + j], ex);
    }
}

// ---------- agg = num/(den+1e-16) + x ; emit bf16 MLP input ----------
__global__ void k_node_agg(const float* __restrict__ num, const float* __restrict__ den,
                           const float* __restrict__ xin, __bf16* __restrict__ out, int n) {
    int i = blockIdx.x * blockDim.x + threadIdx.x;
    if (i >= n) return;
    out[i] = (__bf16)(num[i] / (den[i] + 1e-16f) + xin[i]);
}

// ---------- tiled transpose: fp32 [K,Ncols] -> bf16 column-major [Ncols,K] ----------
__global__ void k_wtranspose(const float* __restrict__ W, __bf16* __restrict__ Wt,
                             int Krows, int Ncols) {
    __shared__ float tile[32][33];
    int kb = blockIdx.x * 32, nb = blockIdx.y * 32;
    int tx = threadIdx.x & 31, ty = threadIdx.x >> 5;     // 256 thr: ty = 0..7
#pragma unroll
    for (int j = ty; j < 32; j += 8)
        tile[j][tx] = W[(size_t)(kb + j) * Ncols + nb + tx];
    __syncthreads();
#pragma unroll
    for (int j = ty; j < 32; j += 8)
        Wt[(size_t)(nb + j) * Krows + kb + tx] = (__bf16)tile[tx][j];
}

// ---------- A fragment loader (ISA 7.12.2 wave32 16x32 bf16 layout) ----------
__device__ __forceinline__ void loadFragA(FragBF& f, const __bf16* row, int k, int koffA) {
    f.q[0] = *(const uint4*)(row + k + koffA);
    f.q[1] = *(const uint4*)(row + k + 16 + koffA);
}

// LDS slot stride for a (colBlock, lane) B-fragment: 32B payload + 16B pad
#define BSLOT 48
#define BSTAGE (128 * BSLOT)   // 4 colBlocks * 32 lanes

// ---------- WMMA bf16 GEMM with LDS-staged, double-buffered B panel ----------
// Block: 256 thr (8 waves) = 128 rows x 64 cols. Wave: 16 rows x 64 cols.
__global__ __launch_bounds__(256, 2)
void k_gemm_bf16(const __bf16* __restrict__ A, const __bf16* __restrict__ Bt,
                 const float* __restrict__ bias, float* __restrict__ C,
                 int Mrows, int Kdim, int Ncols) {
    __shared__ alignas(16) unsigned char lbuf[2][BSTAGE];

    int tid  = threadIdx.x;
    int wave = tid >> 5, lane = tid & 31;
    int colTile = blockIdx.y;                         // 64-col panel
    int nRowTiles = (Mrows + 15) >> 4;
    int rt = blockIdx.x * 8 + wave;                   // 16-row tile per wave
    if (rt > nRowTiles - 1) rt = nRowTiles - 1;       // clamp: duplicate work, benign

    int m     = lane & 15;
    int hi    = lane >> 4;
    int koffA = hi * 8;                               // A: lanes16-31 hold K+8 block

    const __bf16* arow = A + (size_t)(rt * 16 + m) * Kdim;

    // cooperative B staging: thread copies 16B of the 32Kx64N panel per step
    int slot = tid >> 1, half = tid & 1;              // slot = cBlk*32 + srcLane
    int cBlk = slot >> 5, sl = slot & 31;
    const __bf16* bsrc = Bt + (size_t)(colTile * 64 + cBlk * 16 + (sl & 15)) * Kdim
                            + (sl >> 4) * 8 + half * 16;
    unsigned char* d0 = lbuf[0] + slot * BSLOT + half * 16;
    unsigned char* d1 = lbuf[1] + slot * BSLOT + half * 16;

    v8f acc[4] = {v8f{}, v8f{}, v8f{}, v8f{}};
    FragBF faA, faB;

    // prologue: stage K=0 panel, load first A fragment
    *(uint4*)d0 = *(const uint4*)(bsrc + 0);
    loadFragA(faA, arow, 0, koffA);
    __syncthreads();

    for (int k = 0; k < Kdim; k += 64) {
        // stage K=k+32 panel into buf1 while computing on buf0
        *(uint4*)d1 = *(const uint4*)(bsrc + k + 32);
        loadFragA(faB, arow, k + 32, koffA);
#pragma unroll
        for (int c = 0; c < 4; ++c) {
            FragBF fb;
            const unsigned char* pb = lbuf[0] + (size_t)(c * 32 + lane) * BSLOT;
            fb.q[0] = *(const uint4*)(pb);
            fb.q[1] = *(const uint4*)(pb + 16);
            acc[c] = WMMA_BF16(faA.v, fb.v, acc[c]);
        }
        __syncthreads();
        // stage K=k+64 panel into buf0 while computing on buf1
        if (k + 64 < Kdim) {
            *(uint4*)d0 = *(const uint4*)(bsrc + k + 64);
            loadFragA(faA, arow, k + 64, koffA);
        }
#pragma unroll
        for (int c = 0; c < 4; ++c) {
            FragBF fb;
            const unsigned char* pb = lbuf[1] + (size_t)(c * 32 + lane) * BSLOT;
            fb.q[0] = *(const uint4*)(pb);
            fb.q[1] = *(const uint4*)(pb + 16);
            acc[c] = WMMA_BF16(faB.v, fb.v, acc[c]);
        }
        __syncthreads();
    }

    int rbase = hi * 8;                               // C: lanes16-31 hold M+8 rows
#pragma unroll
    for (int c = 0; c < 4; ++c) {
        int col = colTile * 64 + c * 16 + m;
        float bv = bias[col];
        float* cp = C + (size_t)(rt * 16 + rbase) * Ncols + col;
#pragma unroll
        for (int r = 0; r < 8; ++r)
            cp[(size_t)r * Ncols] = acc[c][r] + bv;
    }
}

// ---------- per-row LayerNorm + ReLU -> bf16 (MLP hidden) ----------
__global__ void k_ln_relu_bf16(const float* __restrict__ in, const float* __restrict__ g,
                               const float* __restrict__ b, __bf16* __restrict__ out,
                               int width) {
    __shared__ float sA[256], sB[256];
    int row = blockIdx.x, tid = threadIdx.x;
    const float* rp = in + (size_t)row * width;
    float s = 0.f, ss = 0.f;
    for (int c = tid; c < width; c += 256) { float v = rp[c]; s += v; ss += v * v; }
    sA[tid] = s; sB[tid] = ss; __syncthreads();
    for (int o = 128; o > 0; o >>= 1) {
        if (tid < o) { sA[tid] += sA[tid + o]; sB[tid] += sB[tid + o]; }
        __syncthreads();
    }
    float mu = sA[0] / width;
    float var = sB[0] / width - mu * mu;
    float inv = rsqrtf(var + 1e-5f);
    for (int c = tid; c < width; c += 256) {
        float v = (rp[c] - mu) * inv * g[c] + b[c];
        out[(size_t)row * width + c] = (__bf16)fmaxf(v, 0.f);
    }
}

// ---------- DeepGCN residual: x_out = x_in + relu(LN(y)) ----------
__global__ void k_res_ln_relu(const float* __restrict__ y, const float* __restrict__ g,
                              const float* __restrict__ b, const float* __restrict__ xin,
                              float* __restrict__ xout, int width) {
    __shared__ float sA[256], sB[256];
    int row = blockIdx.x, tid = threadIdx.x;
    const float* rp = y + (size_t)row * width;
    float s = 0.f, ss = 0.f;
    for (int c = tid; c < width; c += 256) { float v = rp[c]; s += v; ss += v * v; }
    sA[tid] = s; sB[tid] = ss; __syncthreads();
    for (int o = 128; o > 0; o >>= 1) {
        if (tid < o) { sA[tid] += sA[tid + o]; sB[tid] += sB[tid + o]; }
        __syncthreads();
    }
    float mu = sA[0] / width;
    float var = sB[0] / width - mu * mu;
    float inv = rsqrtf(var + 1e-5f);
    for (int c = tid; c < width; c += 256) {
        float v = (rp[c] - mu) * inv * g[c] + b[c];
        xout[(size_t)row * width + c] = xin[(size_t)row * width + c] + fmaxf(v, 0.f);
    }
}

// ---------- concat [x1|x2|x3] -> bf16 [N, 3H] ----------
__global__ void k_cat_bf16(const float* __restrict__ x1, const float* __restrict__ x2,
                           const float* __restrict__ x3, __bf16* __restrict__ out, int n) {
    int i = blockIdx.x * blockDim.x + threadIdx.x;
    if (i >= n) return;
    int row = i / H3, c = i % H3;
    float v;
    if (c < Hh)        v = x1[(size_t)row * Hh + c];
    else if (c < H2)   v = x2[(size_t)row * Hh + (c - Hh)];
    else               v = x3[(size_t)row * Hh + (c - H2)];
    out[i] = (__bf16)v;
}

// =====================================================================

static inline void launch_gemm(const __bf16* A, const __bf16* Bt, const float* bias,
                               float* C, int Mrows, int Kdim, int Ncols,
                               hipStream_t stream) {
    int nRowTiles = (Mrows + 15) / 16;
    dim3 grid((nRowTiles + 7) / 8, Ncols / 64);
    k_gemm_bf16<<<grid, 256, 0, stream>>>(A, Bt, bias, C, Mrows, Kdim, Ncols);
}

extern "C" void kernel_launch(void* const* d_in, const int* in_sizes, int n_in,
                              void* d_out, int out_size, void* d_ws, size_t ws_size,
                              hipStream_t stream) {
    const float* x  = (const float*)d_in[0];
    const int*   ei = (const int*)d_in[1];
    const float* t_[3], *w1_[3], *b1_[3], *lg_[3], *lb_[3], *w2_[3], *b2_[3];
    for (int i = 0; i < 3; ++i) {
        t_[i]  = (const float*)d_in[2 + 7 * i];
        w1_[i] = (const float*)d_in[3 + 7 * i];
        b1_[i] = (const float*)d_in[4 + 7 * i];
        lg_[i] = (const float*)d_in[5 + 7 * i];
        lb_[i] = (const float*)d_in[6 + 7 * i];
        w2_[i] = (const float*)d_in[7 + 7 * i];
        b2_[i] = (const float*)d_in[8 + 7 * i];
    }
    const float* lng1 = (const float*)d_in[23];
    const float* lnb1 = (const float*)d_in[24];
    const float* lng2 = (const float*)d_in[25];
    const float* lnb2 = (const float*)d_in[26];
    const float* cw   = (const float*)d_in[27];
    const float* cb   = (const float*)d_in[28];

    // ---- workspace carve-up (time-multiplexed aliasing) ----
    const size_t NH = (size_t)Nn * Hh, N2H = 2 * NH, N3H = 3 * NH;
    char* p = (char*)d_ws;
    auto carve = [&](size_t bytes) { void* r = p; p += (bytes + 255) & ~(size_t)255; return r; };
    float*  x1    = (float*)carve(NH * 4);
    float*  x2    = (float*)carve(NH * 4);
    float*  x3    = (float*)carve(NH * 4);
    float*  R1    = (float*)carve(N2H * 4);   // edge: [zmax | num] ; MLP: mid[N,2H]
    float*  R2    = (float*)carve(NH * 4);    // edge: den          ; MLP: y[N,H]
    __bf16* actbf = (__bf16*)carve(N3H * 2);  // h-bf16 | mid-bf16 ; reused as concat
    __bf16* wbf   = (__bf16*)carve((size_t)3932160 * 2);

    __bf16* w1t[3] = { wbf, wbf + 524288, wbf + 1048576 };
    __bf16* w2t[3] = { wbf + 1572864, wbf + 2097152, wbf + 2621440 };
    __bf16* cwt    = wbf + 3145728;

    // ---- weights -> bf16, transposed to column-major (tiled, coalesced) ----
    for (int i = 0; i < 3; ++i) {
        k_wtranspose<<<dim3(Hh / 32, H2 / 32), 256, 0, stream>>>(w1_[i], w1t[i], Hh, H2);
        k_wtranspose<<<dim3(H2 / 32, Hh / 32), 256, 0, stream>>>(w2_[i], w2t[i], H2, Hh);
    }
    k_wtranspose<<<dim3(H3 / 32, Hh / 32), 256, 0, stream>>>(cw, cwt, H3, Hh);

    const int nhBlocks = (int)((NH + 255) / 256);
    const int eThreads = Ee * (Hh / 4);
    const int eBlocks  = (eThreads + 255) / 256;

    auto run_conv = [&](const float* xin, int li, float* y_target) {
        unsigned* zmax = (unsigned*)R1;
        float* num = R1 + NH;
        float* den = R2;
        k_init_seg<<<nhBlocks, 256, 0, stream>>>(zmax, num, den, (int)NH);
        k_edge_max<<<eBlocks, 256, 0, stream>>>((const float4*)xin, ei, t_[li], zmax, Ee, Hh / 4);
        k_zmax_fix<<<nhBlocks, 256, 0, stream>>>(zmax, (int)NH);
        k_edge_sum<<<eBlocks, 256, 0, stream>>>((const float4*)xin, ei, t_[li],
                                                (const float*)zmax, num, den, Ee, Hh / 4);
        __bf16* habf = actbf;
        k_node_agg<<<nhBlocks, 256, 0, stream>>>(num, den, xin, habf, (int)NH);
        float* mid = R1;                          // overwrites zmax/num (already consumed)
        launch_gemm(habf, w1t[li], b1_[li], mid, Nn, Hh, H2, stream);
        __bf16* midbf = actbf + NH;
        k_ln_relu_bf16<<<Nn, 256, 0, stream>>>(mid, lg_[li], lb_[li], midbf, H2);
        launch_gemm(midbf, w2t[li], b2_[li], y_target, Nn, H2, Hh, stream);
    };

    // layer 0: x1 = conv0(x)
    run_conv(x, 0, x1);
    // layer 1: x2 = x1 + relu(LN1(conv1(x1)))
    run_conv(x1, 1, R2);
    k_res_ln_relu<<<Nn, 256, 0, stream>>>(R2, lng1, lnb1, x1, x2, Hh);
    // layer 2: x3 = x2 + relu(LN2(conv2(x2)))
    run_conv(x2, 2, R2);
    k_res_ln_relu<<<Nn, 256, 0, stream>>>(R2, lng2, lnb2, x2, x3, Hh);

    // out = concat(x1,x2,x3) @ cw + cb
    k_cat_bf16<<<(int)((N3H + 255) / 256), 256, 0, stream>>>(x1, x2, x3, actbf, (int)N3H);
    launch_gemm(actbf, cwt, cb, (float*)d_out, Nn, H3, Hh, stream);
}
